// ExaoneMoEAttention_26620207301233
// MI455X (gfx1250) — matrix operand order repaired and checked by reading, not compiled, not run on hardware
//
#include <hip/hip_runtime.h>
#include <hip/hip_bf16.h>

#define T_    2048
#define HID_  4096
#define NH    32
#define NKV   8
#define HD    128
#define QKV_N ((NH + 2*NKV)*HD)   // 6144
#define WIN   1024
#define EPSF  1e-5f
#define THETA_ 1000000.0f

typedef __bf16 bf16;
typedef bf16  v8bf  __attribute__((ext_vector_type(8)));
typedef bf16  v16bf __attribute__((ext_vector_type(16)));
typedef float v8f   __attribute__((ext_vector_type(8)));

__device__ __forceinline__ bf16 f2bf(float f) {
  unsigned u = __builtin_bit_cast(unsigned, f);
  unsigned r = u + 0x7FFFu + ((u >> 16) & 1u);   // round-to-nearest-even
  unsigned short h = (unsigned short)(r >> 16);
  return __builtin_bit_cast(bf16, h);
}
__device__ __forceinline__ v8bf ld8(const bf16* p) { return *(const v8bf*)p; }
__device__ __forceinline__ v16bf cat16(v8bf lo, v8bf hi) {
  return __builtin_shufflevector(lo, hi, 0,1,2,3,4,5,6,7,8,9,10,11,12,13,14,15);
}

// ---------------- cast / transpose-cast ----------------
__global__ void cast_kernel(const float* __restrict__ src, bf16* __restrict__ dst, int n) {
  int i = blockIdx.x * blockDim.x + threadIdx.x;
  if (i < n) dst[i] = f2bf(src[i]);
}
__global__ void transpose_cast_kernel(const float* __restrict__ src, bf16* __restrict__ dst,
                                      int R, int C) {
  int i = blockIdx.x * blockDim.x + threadIdx.x;
  if (i < R * C) {
    int r = i / C, c = i % C;
    dst[(long)c * R + r] = f2bf(src[i]);   // dst is C x R row-major (i.e. src^T)
  }
}

// ---------------- bf16 WMMA GEMM: C(MxN,f32) = A(MxK) * Bt(NxK)^T ----------------
// Register-blocked: each wave computes a 32x64 tile (2 A-frags x 4 B-frags, 8 accums).
__global__ __launch_bounds__(256) void gemm_bf16_kernel(
    const bf16* __restrict__ A, const bf16* __restrict__ Bt, float* __restrict__ C,
    int M, int N, int K) {
  int wave = blockIdx.x * (blockDim.x >> 5) + (threadIdx.x >> 5);
  int tiles_n = N >> 6;                  // 64-wide tiles
  int tiles_m = M >> 5;                  // 32-tall tiles
  if (wave >= tiles_m * tiles_n) return;
  int m0 = (wave / tiles_n) << 5;
  int n0 = (wave % tiles_n) << 6;
  int lane = threadIdx.x & 31;
  int l15 = lane & 15, hi = lane >> 4;

  const bf16* arow0 = A + (long)(m0 + l15) * K;        // rows m0..m0+15
  const bf16* arow1 = arow0 + (long)16 * K;            // rows m0+16..m0+31
  const bf16* brow  = Bt + (long)(n0 + l15) * K;       // cols n0..n0+63 (stride 16*K per subtile)

  v8f acc[2][4];
  const v8f vz = {};
#pragma unroll
  for (int i = 0; i < 2; ++i)
#pragma unroll
    for (int j = 0; j < 4; ++j) acc[i][j] = vz;

  for (int k = 0; k < K; k += 32) {
    // A 16x32 fragments: lanes 0-15 K={0..7,16..23}, lanes 16-31 K={8..15,24..31}
    v16bf a0 = cat16(ld8(arow0 + k + hi * 8), ld8(arow0 + k + 16 + hi * 8));
    v16bf a1 = cat16(ld8(arow1 + k + hi * 8), ld8(arow1 + k + 16 + hi * 8));
#pragma unroll
    for (int j = 0; j < 4; ++j) {
      // B 32x16 fragment: lanes 0-15 K={0..7,16..23}, lanes 16-31 K={8..15,24..31} (col = n0+16j+l15)
      const bf16* bp = brow + (long)j * 16 * K + k + hi * 8;
      v16bf b = cat16(ld8(bp), ld8(bp + 16));
      acc[0][j] = __builtin_amdgcn_wmma_f32_16x16x32_bf16(false, a0, false, b, (short)0, acc[0][j], false, false);
      acc[1][j] = __builtin_amdgcn_wmma_f32_16x16x32_bf16(false, a1, false, b, (short)0, acc[1][j], false, false);
    }
  }
#pragma unroll
  for (int i = 0; i < 2; ++i) {
#pragma unroll
    for (int j = 0; j < 4; ++j) {
      float* crow = C + (long)(m0 + i * 16 + (hi << 3)) * N + n0 + j * 16 + l15;
#pragma unroll
      for (int r = 0; r < 8; ++r) crow[(long)r * N] = acc[i][j][r];
    }
  }
}

// ---------------- RMSNorm + RoPE + layout shuffle ----------------
__global__ __launch_bounds__(128) void postproc_kernel(
    const float* __restrict__ qkv, const int* __restrict__ positions,
    const float* __restrict__ qnw, const float* __restrict__ knw,
    bf16* __restrict__ qb,   // [NH][T][HD], scale folded in
    bf16* __restrict__ kb,   // [NKV][T][HD]
    bf16* __restrict__ vT) { // [NKV][HD][T]
  __shared__ float red[128];
  __shared__ float sh[128];
  int t = blockIdx.x;
  int d = threadIdx.x;
  float pos = (float)positions[t];
  const float scale = 0.08838834764831845f;   // 128^-0.5
  float cg = 0.f, sg = 0.f;
  if (d < 64) {
    float inv = __powf(THETA_, -((float)d) / 64.0f);
    __sincosf(pos * inv, &sg, &cg);
  }
  const float* row = qkv + (long)t * QKV_N;
  for (int h = 0; h < NH + NKV; ++h) {
    bool isq = h < NH;
    int off = isq ? h * HD : NH * HD + (h - NH) * HD;
    float x = row[off + d];
    red[d] = x * x;
    __syncthreads();
    for (int st = 64; st > 0; st >>= 1) { if (d < st) red[d] += red[d + st]; __syncthreads(); }
    float inv = rsqrtf(red[0] / 128.0f + EPSF);
    float w = isq ? qnw[d] : knw[d];
    sh[d] = x * inv * w;
    __syncthreads();
    if (d < 64) {
      float x1 = sh[d], x2 = sh[d + 64];
      float o1 = x1 * cg - x2 * sg;
      float o2 = x2 * cg + x1 * sg;
      if (isq) {
        bf16* q = qb + ((long)h * T_ + t) * HD;
        q[d] = f2bf(o1 * scale); q[d + 64] = f2bf(o2 * scale);
      } else {
        bf16* kk = kb + ((long)(h - NH) * T_ + t) * HD;
        kk[d] = f2bf(o1); kk[d + 64] = f2bf(o2);
      }
    }
    __syncthreads();
  }
  for (int kh = 0; kh < NKV; ++kh) {
    float x = row[(NH + NKV) * HD + kh * HD + d];
    vT[((long)kh * HD + d) * T_ + t] = f2bf(x);
  }
}

// ---------------- flash-style sliding-window attention (1 wave / 16q-tile / head) ----------------
__global__ __launch_bounds__(32) void attn_kernel(
    const bf16* __restrict__ qb, const bf16* __restrict__ kb,
    const bf16* __restrict__ vT, bf16* __restrict__ attnb) {
  __shared__ bf16 P[16][32];
  int h  = blockIdx.x & (NH - 1);
  int qt = blockIdx.x / NH;
  int q0 = qt << 4;
  int kh = h >> 2;                       // rep = NH/NKV = 4
  int lane = threadIdx.x & 31;
  int l15 = lane & 15, hi = lane >> 4;

  // preload Q A-fragments: 4 chunks of K=32 across D=128
  const bf16* qrow = qb + ((long)h * T_ + (q0 + l15)) * HD;
  v16bf qf[4];
#pragma unroll
  for (int i = 0; i < 4; ++i) {
    int kd = i * 32 + hi * 16;
    qf[i] = cat16(ld8(qrow + kd), ld8(qrow + kd + 8));
  }

  float mrow[8], lrow[8], alph[8];
  v8f o[8];
  const v8f vz = {};
#pragma unroll
  for (int r = 0; r < 8; ++r) { mrow[r] = -INFINITY; lrow[r] = 0.f; o[r] = vz; }

  int smin = q0 - (WIN - 1); if (smin < 0) smin = 0;
  int sc0 = smin & ~31;
  int send = q0 + 15;

  for (int sc = sc0; sc <= send; sc += 32) {
    v8f s0 = {}, s1 = {};
    const bf16* k0 = kb + ((long)kh * T_ + (sc + l15)) * HD;
    const bf16* k1 = k0 + 16 * HD;
#pragma unroll
    for (int i = 0; i < 4; ++i) {
      int kd = i * 32 + hi * 16;
      v16bf b0 = cat16(ld8(k0 + kd), ld8(k0 + kd + 8));
      s0 = __builtin_amdgcn_wmma_f32_16x16x32_bf16(false, qf[i], false, b0, (short)0, s0, false, false);
      v16bf b1 = cat16(ld8(k1 + kd), ld8(k1 + kd + 8));
      s1 = __builtin_amdgcn_wmma_f32_16x16x32_bf16(false, qf[i], false, b1, (short)0, s1, false, false);
    }
    int col0 = sc + l15, col1 = col0 + 16;
#pragma unroll
    for (int r = 0; r < 8; ++r) {
      int trow = q0 + (hi << 3) + r;
      bool v0 = (col0 <= trow) && (trow - col0 < WIN);
      bool v1 = (col1 <= trow) && (trow - col1 < WIN);
      float e0 = v0 ? s0[r] : -INFINITY;
      float e1 = v1 ? s1[r] : -INFINITY;
      float cm = fmaxf(e0, e1);
      for (int m = 8; m >= 1; m >>= 1) cm = fmaxf(cm, __shfl_xor(cm, m, 16));
      float mn = fmaxf(mrow[r], cm);
      float al = (mrow[r] == mn) ? 1.0f : __expf(mrow[r] - mn);
      float p0 = v0 ? __expf(s0[r] - mn) : 0.f;
      float p1 = v1 ? __expf(s1[r] - mn) : 0.f;
      float rs = p0 + p1;
      for (int m = 8; m >= 1; m >>= 1) rs += __shfl_xor(rs, m, 16);
      lrow[r] = lrow[r] * al + rs;
      mrow[r] = mn;
      alph[r] = al;
      int prow = (hi << 3) + r;
      P[prow][l15]      = f2bf(p0);
      P[prow][16 + l15] = f2bf(p1);
    }
#pragma unroll
    for (int f = 0; f < 8; ++f)
#pragma unroll
      for (int r = 0; r < 8; ++r) o[f][r] *= alph[r];
    __syncthreads();
    // C-layout -> A-layout for P, then O += P * V
    v16bf ap = cat16(ld8(&P[l15][hi * 16]), ld8(&P[l15][hi * 16 + 8]));
#pragma unroll
    for (int f = 0; f < 8; ++f) {
      const bf16* vrow = vT + ((long)kh * HD + (f * 16 + l15)) * T_ + sc + hi * 16;
      v16bf bv = cat16(ld8(vrow), ld8(vrow + 8));
      o[f] = __builtin_amdgcn_wmma_f32_16x16x32_bf16(false, ap, false, bv, (short)0, o[f], false, false);
    }
    __syncthreads();
  }

#pragma unroll
  for (int r = 0; r < 8; ++r) {
    int trow = q0 + (hi << 3) + r;
    float invl = 1.0f / lrow[r];
    bf16* orow = attnb + (long)trow * (NH * HD) + h * HD + l15;
#pragma unroll
    for (int f = 0; f < 8; ++f) orow[f * 16] = f2bf(o[f][r] * invl);
  }
}

// ---------------- launcher ----------------
extern "C" void kernel_launch(void* const* d_in, const int* in_sizes, int n_in,
                              void* d_out, int out_size, void* d_ws, size_t ws_size,
                              hipStream_t stream) {
  const int*   positions = (const int*)d_in[0];
  const float* hs   = (const float*)d_in[1];
  const float* wqkv = (const float*)d_in[2];
  const float* qnw  = (const float*)d_in[3];
  const float* knw  = (const float*)d_in[4];
  const float* wo   = (const float*)d_in[5];
  float* out = (float*)d_out;

  char* p = (char*)d_ws;
  auto alloc = [&](size_t bytes) { char* r = p; p += (bytes + 255) & ~(size_t)255; return r; };
  bf16*  hsb   = (bf16*) alloc((size_t)T_ * HID_ * 2);
  bf16*  wqkvT = (bf16*) alloc((size_t)QKV_N * HID_ * 2);
  bf16*  woT   = (bf16*) alloc((size_t)HID_ * (NH * HD) * 2);
  float* qkv   = (float*)alloc((size_t)T_ * QKV_N * 4);
  bf16*  qb    = (bf16*) alloc((size_t)NH  * T_ * HD * 2);
  bf16*  kb    = (bf16*) alloc((size_t)NKV * T_ * HD * 2);
  bf16*  vT    = (bf16*) alloc((size_t)NKV * HD * T_ * 2);
  bf16*  attnb = (bf16*) alloc((size_t)T_ * NH * HD * 2);

  int n1 = T_ * HID_;
  cast_kernel<<<(n1 + 255) / 256, 256, 0, stream>>>(hs, hsb, n1);
  int n2 = HID_ * QKV_N;
  transpose_cast_kernel<<<(n2 + 255) / 256, 256, 0, stream>>>(wqkv, wqkvT, HID_, QKV_N);
  int n3 = (NH * HD) * HID_;
  transpose_cast_kernel<<<(n3 + 255) / 256, 256, 0, stream>>>(wo, woT, NH * HD, HID_);

  { int tiles = (T_ / 32) * (QKV_N / 64);          // 32x64 per wave
    gemm_bf16_kernel<<<(tiles + 7) / 8, 256, 0, stream>>>(hsb, wqkvT, qkv, T_, QKV_N, HID_); }

  postproc_kernel<<<T_, 128, 0, stream>>>(qkv, positions, qnw, knw, qb, kb, vT);

  attn_kernel<<<(T_ / 16) * NH, 32, 0, stream>>>(qb, kb, vT, attnb);

  { int tiles = (T_ / 32) * (HID_ / 64);
    gemm_bf16_kernel<<<(tiles + 7) / 8, 256, 0, stream>>>(attnb, woT, out, T_, HID_, HID_); }
}
